// DecoderPixelCNNppBlock_10814727651740
// MI455X (gfx1250) — matrix operand order repaired and checked by reading, not compiled, not run on hardware
//
#include <hip/hip_runtime.h>
#include <hip/hip_bf16.h>

// ---------------------------------------------------------------------------
// PixelCNN++ decoder block for MI455X (gfx1250, wave32, WMMA bf16)
//   S=4 B=32 -> N=128 images, C=1, F=64, HW=28, D=64
// Core: implicit-GEMM conv on v_wmma_f32_16x16x32_bf16, concat_elu fused into
// the im2col fetch, fp32 accumulate, elementwise epilogues on VALU.
// Round 2: 3-D grid (no divide prologue), division-free incremental im2col
// staging with coalesced activation gathers.
// ---------------------------------------------------------------------------

#define NIMG 128
#define FCH  64

typedef __bf16 bf16_t;
typedef __attribute__((ext_vector_type(16))) bf16_t bf16x16;
typedef __attribute__((ext_vector_type(8)))  float  f32x8;
typedef __attribute__((ext_vector_type(4)))  unsigned int u32x4;

union BFrag { u32x4 q[2]; bf16x16 v; };

__device__ __forceinline__ f32x8 wmma_bf16(bf16x16 a, bf16x16 b, f32x8 c) {
    // v_wmma_f32_16x16x32_bf16 : D = A(16x32) * B(32x16) + C(16x16 f32)
    return __builtin_amdgcn_wmma_f32_16x16x32_bf16(
        /*neg_a=*/false, a, /*neg_b=*/false, b,
        /*c_mod=*/(short)0, c, /*reuse_a=*/false, /*reuse_b=*/false);
}

__device__ __forceinline__ unsigned short f2bf(float f) {
    unsigned int u = __float_as_uint(f);
    unsigned int r = u + 0x7FFFu + ((u >> 16) & 1u);   // round-to-nearest-even
    return (unsigned short)(r >> 16);
}

__device__ __forceinline__ float elu1(float x) {
    return x > 0.f ? x : (__expf(x) - 1.f);
}

// ---------------------------------------------------------------------------
// Generic implicit-GEMM convolution descriptor
// ---------------------------------------------------------------------------
struct ConvDesc {
    const float* x0;    // physical input part 0 (c0 channels)
    const float* x1;    // physical input part 1 (c1n channels) or null
    const float* wgt;   // weights
    const float* bias;  // bias[Co]
    float* y;           // output (NCHW)
    int c0, c1n;        // physical channel counts
    int celu;           // 1 => logical channels = 2*(c0+c1n), elu(+x / -x)
    int xpmode;         // 1 => input = [x_in[(n&31)], ones] (2 channels, 28x28)
    int wtrans;         // 1 => weights are (K, Co) row-major (nin layout)
    int H, W, OH, OW;
    int Co;             // output channels (multiple of 64)
    int K;              // CiLogical * fh * fw
    int fh, fw, pt, pl; // filter + top/left pad (shifts folded in)
    int sh, sw;         // stride
    int accum;          // 1 => y += result (bias still added)
    int zr0, zc0;       // zero contribution at output row 0 / col 0
};

__device__ __forceinline__ float conv_fetch(const ConvDesc& d, int n, int ci,
                                            int ih, int iw) {
    if (d.xpmode) {
        if (ci >= 1) return 1.0f;
        return d.x0[(size_t)(n & 31) * (28 * 28) + ih * 28 + iw];
    }
    int CiPhys = d.c0 + d.c1n;
    int cc = ci;
    float s = 1.f;
    if (d.celu && ci >= CiPhys) { cc = ci - CiPhys; s = -1.f; }
    float v;
    if (cc < d.c0)
        v = d.x0[((size_t)n * d.c0 + cc) * (d.H * d.W) + ih * d.W + iw];
    else
        v = d.x1[((size_t)n * d.c1n + (cc - d.c0)) * (d.H * d.W) + ih * d.W + iw];
    v *= s;
    if (d.celu) v = elu1(v);
    return v;
}

#define KC   128   // K chunk through LDS
#define KPAD 136   // row stride in bf16 (pad 16B -> conflict-free b128 reads)

// grid: (n, pixTile, coTile); block: 128 threads = 4 waves; each wave owns a
// 16-co stripe x 64 pixels (4 WMMA accumulators).
__global__ __launch_bounds__(128) void conv_wmma_kernel(ConvDesc d) {
    __shared__ alignas(16) unsigned short lwa[64 * KPAD]; // A: weights [co][k]
    __shared__ alignas(16) unsigned short lwb[64 * KPAD]; // B: pixels  [px][k]

    const int OHW = d.OH * d.OW;
    const int n       = blockIdx.x;
    const int pixBase = blockIdx.y << 6;
    const int coBase  = blockIdx.z << 6;

    const int tid  = threadIdx.x;
    const int lane = tid & 31;
    const int wave = tid >> 5;
    const int half = lane >> 4;
    const int l15  = lane & 15;
    const int khw  = d.fh * d.fw;

    // per-thread pixel ownership for im2col staging: 2 threads per pixel,
    // thread covers a contiguous 64-wide K span -> coalesced gathers
    const int spx = tid >> 1;
    const int sks = (tid & 1) << 6;
    const int sp  = pixBase + spx;
    int soh = 0, sow = 0;
    const bool spValid = (sp < OHW);
    if (spValid) { soh = sp / d.OW; sow = sp - soh * d.OW; }

    f32x8 acc[4];
    #pragma unroll
    for (int t = 0; t < 4; ++t) acc[t] = (f32x8)0.f;

    const int nChunks = (d.K + KC - 1) / KC;
    for (int ch = 0; ch < nChunks; ++ch) {
        const int kc = ch * KC;
        // prefetch next weight chunk into L2 (global_prefetch_b8)
        if (ch + 1 < nChunks && tid == 0 && !d.wtrans)
            __builtin_prefetch(d.wgt + (size_t)coBase * d.K + kc + KC, 0, 1);

        // ---- stage weights: 64 co x 128 k -> bf16 LDS (coalesced) ----
        for (int i = tid; i < 64 * KC; i += 128) {
            int co = i >> 7, k = i & (KC - 1), kk = kc + k;
            float v = 0.f;
            if (kk < d.K) {
                if (d.wtrans) v = d.wgt[(size_t)kk * d.Co + coBase + co];
                else          v = d.wgt[(size_t)(coBase + co) * d.K + kk];
            }
            lwa[co * KPAD + k] = f2bf(v);
        }

        // ---- stage im2col input: division-free incremental (ci,kh,kw) ----
        {
            int kk0 = kc + sks;
            int ci = kk0 / khw;
            int r2 = kk0 - ci * khw;
            int kh = r2 / d.fw;
            int kw = r2 - kh * d.fw;
            unsigned short* dst = &lwb[spx * KPAD + sks];
            #pragma unroll 4
            for (int j = 0; j < 64; ++j) {
                float v = 0.f;
                if (spValid && (kk0 + j) < d.K) {
                    int ih = soh * d.sh - d.pt + kh;
                    int iw = sow * d.sw - d.pl + kw;
                    if (ih >= 0 && ih < d.H && iw >= 0 && iw < d.W)
                        v = conv_fetch(d, n, ci, ih, iw);
                }
                dst[j] = f2bf(v);
                if (++kw == d.fw) { kw = 0; if (++kh == d.fh) { kh = 0; ++ci; } }
            }
        }
        __syncthreads();

        // ---- 16 WMMAs per chunk per wave ----
        const int arow = (wave << 4) + l15; // A row (co within 64-tile)
        #pragma unroll
        for (int kk = 0; kk < 4; ++kk) {
            BFrag af;
            // A 16x32 bf16: VGPR0-3 = K(half*8 .. +7), VGPR4-7 = K(16+half*8 ..)
            af.q[0] = *(const u32x4*)&lwa[arow * KPAD + kk * 32 + half * 8];
            af.q[1] = *(const u32x4*)&lwa[arow * KPAD + kk * 32 + 16 + half * 8];
            #pragma unroll
            for (int nt = 0; nt < 4; ++nt) {
                int brow = (nt << 4) + l15; // pixel within 64-tile
                BFrag bf;
                // B 32x16 bf16: VGPR0-3 = K(half*8 .. +7), VGPR4-7 = K(16+half*8 ..)
                bf.q[0] = *(const u32x4*)&lwb[brow * KPAD + kk * 32 + half * 8];
                bf.q[1] = *(const u32x4*)&lwb[brow * KPAD + kk * 32 + 16 + half * 8];
                acc[nt] = wmma_bf16(af.v, bf.v, acc[nt]);
            }
        }
        __syncthreads();
    }

    // ---- epilogue: C/D layout -> n = lane&15, m = r + half*8 ----
    #pragma unroll
    for (int nt = 0; nt < 4; ++nt) {
        int p = pixBase + (nt << 4) + l15;
        if (p >= OHW) continue;
        int oh = p / d.OW, ow = p - oh * d.OW;
        #pragma unroll
        for (int r = 0; r < 8; ++r) {
            int co = coBase + (wave << 4) + r + half * 8;
            float v = acc[nt][r] + d.bias[co];
            if (d.zr0 && oh == 0) v = 0.f;
            if (d.zc0 && ow == 0) v = 0.f;
            size_t off = ((size_t)n * d.Co + co) * OHW + p;
            if (d.accum) v += d.y[off];
            d.y[off] = v;
        }
    }
}

// ---------------------------------------------------------------------------
// h-projection: hb[n, 2F] = z[n,:] @ W(D x 2F) + b
// ---------------------------------------------------------------------------
__global__ void hproj_kernel(const float* __restrict__ z,
                             const float* __restrict__ w,
                             const float* __restrict__ b,
                             float* __restrict__ hb) {
    int idx = blockIdx.x * blockDim.x + threadIdx.x; // 128*128
    if (idx >= NIMG * 128) return;
    int co = idx & 127, n = idx >> 7;
    float acc = b[co];
    #pragma unroll 8
    for (int dd = 0; dd < 64; ++dd) acc += z[n * 64 + dd] * w[dd * 128 + co];
    hb[idx] = acc;
}

// gated epilogue: out = x + (c[f]+hb[f]) * sigmoid(c[f+F]+hb[f+F])
__global__ void gate_kernel(const float* __restrict__ x,
                            const float* __restrict__ c,
                            const float* __restrict__ hb,
                            float* __restrict__ out, int HWn) {
    int idx = blockIdx.x * blockDim.x + threadIdx.x;
    int total = NIMG * FCH * HWn;
    if (idx >= total) return;
    int p = idx % HWn;
    int f = (idx / HWn) & (FCH - 1);
    int n = idx / (HWn * FCH);
    float g1 = c[((size_t)n * 128 + f) * HWn + p] + hb[n * 128 + f];
    float g2 = c[((size_t)n * 128 + FCH + f) * HWn + p] + hb[n * 128 + FCH + f];
    out[idx] = x[idx] + g1 * (1.f / (1.f + __expf(-g2)));
}

// conv_transpose stride 2 'VALID', HWIO weights; fw==3 output cropped [:,1:]
__global__ void deconv_kernel(const float* __restrict__ x,
                              const float* __restrict__ w,
                              const float* __restrict__ b,
                              float* __restrict__ y,
                              int inH, int inW, int OH, int OW, int fw) {
    int idx = blockIdx.x * blockDim.x + threadIdx.x;
    int total = NIMG * FCH * OH * OW;
    if (idx >= total) return;
    int ow = idx % OW;
    int oh = (idx / OW) % OH;
    int co = (idx / (OW * OH)) & (FCH - 1);
    int n  = idx / (OW * OH * FCH);
    int owp = ow + (fw == 3 ? 1 : 0);
    float acc = b[co];
    for (int kh = 0; kh < 2; ++kh) {
        int ih2 = oh - kh;
        if (ih2 < 0 || (ih2 & 1)) continue;
        int ih = ih2 >> 1;
        if (ih >= inH) continue;
        for (int kw = 0; kw < fw; ++kw) {
            int iw2 = owp - kw;
            if (iw2 < 0 || (iw2 & 1)) continue;
            int iw = iw2 >> 1;
            if (iw >= inW) continue;
            const float* xp = &x[((size_t)n * FCH) * inH * inW + ih * inW + iw];
            const float* wp = &w[((size_t)(kh * fw + kw) * FCH) * FCH + co];
            #pragma unroll 8
            for (int ci = 0; ci < FCH; ++ci)
                acc += xp[(size_t)ci * inH * inW] * wp[(size_t)ci * FCH];
        }
    }
    y[idx] = acc;
}

// final: out[n,0,h,w] = sum_ci elu(ul[n,ci,h,w]) * w[ci] + b[0]
__global__ void final_kernel(const float* __restrict__ ul,
                             const float* __restrict__ w,
                             const float* __restrict__ b,
                             float* __restrict__ out) {
    int idx = blockIdx.x * blockDim.x + threadIdx.x; // 128*784
    if (idx >= NIMG * 784) return;
    int p = idx % 784, n = idx / 784;
    float acc = b[0];
    #pragma unroll 8
    for (int ci = 0; ci < FCH; ++ci)
        acc += elu1(ul[((size_t)n * FCH + ci) * 784 + p]) * w[ci];
    out[idx] = acc;
}

// ---------------------------------------------------------------------------
// Host orchestration
// ---------------------------------------------------------------------------
struct CP { const float* b; const float* w; };
struct GR { CP c1, c2, h, s; };

extern "C" void kernel_launch(void* const* d_in, const int* in_sizes, int n_in,
                              void* d_out, int out_size, void* d_ws, size_t ws_size,
                              hipStream_t stream) {
    const float* z2  = (const float*)d_in[0];  // (128, 64)
    const float* xin = (const float*)d_in[1];  // (32, 1, 28, 28)

    // ---- param unpack (jax tree_flatten: sorted dict keys; b before w) ----
    int pi = 2;
    auto NP  = [&]() { return (const float*)d_in[pi++]; };
    auto cp  = [&]() { CP c; c.b = NP(); c.w = NP(); return c; };
    auto grp = [&](bool has_s) {
        GR g; g.c1 = cp(); g.c2 = cp(); g.h = cp();
        if (has_s) g.s = cp(); else g.s = CP{nullptr, nullptr};
        return g;
    };
    const int NRDN[3] = {2, 3, 3};
    GR dn_u[3][3], dn_ul[3][3], up_u[3][2], up_ul[3][2];
    // 'dn'
    for (int l = 0; l < 3; ++l) {
        for (int r = 0; r < NRDN[l]; ++r) dn_u[l][r]  = grp(true);
        for (int r = 0; r < NRDN[l]; ++r) dn_ul[l][r] = grp(true);
    }
    CP dsz_u[2];  for (int i = 0; i < 2; ++i) dsz_u[i]  = cp();
    CP dsz_ul[2]; for (int i = 0; i < 2; ++i) dsz_ul[i] = cp();
    CP outP   = cp();          // 'out'
    CP u_init = cp();          // 'u_init'
    CP ul_a   = cp();          // 'ul_init_a'
    CP ul_b   = cp();          // 'ul_init_b'
    for (int l = 0; l < 3; ++l) {  // 'up'
        for (int r = 0; r < 2; ++r) up_u[l][r]  = grp(false);
        for (int r = 0; r < 2; ++r) up_ul[l][r] = grp(true);
    }
    CP usz_u[2];  for (int i = 0; i < 2; ++i) usz_u[i]  = cp();
    CP usz_ul[2]; for (int i = 0; i < 2; ++i) usz_ul[i] = cp();

    // ---- workspace ----
    const size_t SZ28 = (size_t)NIMG * FCH * 784;
    const size_t SZ14 = (size_t)NIMG * FCH * 196;
    const size_t SZ7  = (size_t)NIMG * FCH * 49;
    float* W0 = (float*)d_ws;
    size_t off = 0;
    auto alloc = [&](size_t n) { float* p = W0 + off; off += n; return p; };
    const size_t ss[9] = {SZ28, SZ28, SZ28, SZ14, SZ14, SZ14, SZ7, SZ7, SZ7};
    float* stU[9];  for (int i = 0; i < 9; ++i) stU[i]  = alloc(ss[i]);
    float* stUL[9]; for (int i = 0; i < 9; ++i) stUL[i] = alloc(ss[i]);
    float* wU[2]  = {alloc(SZ28), alloc(SZ28)};
    float* wUL[2] = {alloc(SZ28), alloc(SZ28)};
    float* ctmp  = alloc(2 * SZ28);  // 128-channel conv2 output
    float* c1tmp = alloc(SZ28);      // 64-channel conv1 output
    float* hb    = alloc(NIMG * 128);

    // ---- launch helpers ----
    auto conv = [&](const float* wgt, int wtrans, const float* bias,
                    const float* x0, const float* x1, int c0, int c1n,
                    int celu, int xpmode, float* y, int Co, int K,
                    int fh, int fw, int pt, int pl,
                    int H, int Wd, int OH, int OW, int sh, int sw,
                    int accum, int zr0, int zc0) {
        ConvDesc d;
        d.x0 = x0; d.x1 = x1; d.wgt = wgt; d.bias = bias; d.y = y;
        d.c0 = c0; d.c1n = c1n; d.celu = celu; d.xpmode = xpmode; d.wtrans = wtrans;
        d.H = H; d.W = Wd; d.OH = OH; d.OW = OW; d.Co = Co; d.K = K;
        d.fh = fh; d.fw = fw; d.pt = pt; d.pl = pl; d.sh = sh; d.sw = sw;
        d.accum = accum; d.zr0 = zr0; d.zc0 = zc0;
        int pixT = (OH * OW + 63) / 64, coT = Co / 64;
        conv_wmma_kernel<<<dim3(NIMG, pixT, coT), 128, 0, stream>>>(d);
    };
    auto ew = [&](int total) { return dim3((total + 255) / 256); };

    auto gated = [&](const float* x, const GR& g, int drk /*0:ds(2,3) 1:dr(2,2)*/,
                     const float* a0, const float* a1, int Hs, float* outbuf) {
        int fw = drk ? 2 : 3;
        int K1 = 128 * 2 * fw;
        int HWn = Hs * Hs;
        // c1(concat_elu(x))
        conv(g.c1.w, 0, g.c1.b, x, nullptr, FCH, 0, 1, 0, c1tmp, 64, K1,
             2, fw, 1, 1, Hs, Hs, Hs, Hs, 1, 1, 0, 0, 0);
        // + nin(concat_elu(a))
        if (a0) {
            int CiL = a1 ? 256 : 128;
            conv(g.s.w, 1, g.s.b, a0, a1, FCH, a1 ? FCH : 0, 1, 0, c1tmp, 64,
                 CiL, 1, 1, 0, 0, Hs, Hs, Hs, Hs, 1, 1, 1, 0, 0);
        }
        // c2(concat_elu(c1))
        conv(g.c2.w, 0, g.c2.b, c1tmp, nullptr, FCH, 0, 1, 0, ctmp, 128, K1,
             2, fw, 1, 1, Hs, Hs, Hs, Hs, 1, 1, 0, 0, 0);
        hproj_kernel<<<ew(NIMG * 128), 256, 0, stream>>>(z2, g.h.w, g.h.b, hb);
        gate_kernel<<<ew(NIMG * FCH * HWn), 256, 0, stream>>>(x, ctmp, hb, outbuf, HWn);
    };

    // ---- init convs (shifts folded into padding; zr0/zc0 emulate shift zeros)
    // u = down_shift(ds_conv(xp, u_init))        2x3, pt=2, pl=1
    conv(u_init.w, 0, u_init.b, xin, nullptr, 2, 0, 0, 1, stU[0], 64, 12,
         2, 3, 2, 1, 28, 28, 28, 28, 1, 1, 0, 1, 0);
    // ul = down_shift(ds_conv(xp, ul_a))         1x3, pt=1, pl=1
    conv(ul_a.w, 0, ul_a.b, xin, nullptr, 2, 0, 0, 1, stUL[0], 64, 6,
         1, 3, 1, 1, 28, 28, 28, 28, 1, 1, 0, 1, 0);
    //     + right_shift(dr_conv(xp, ul_b))       2x1, pt=1, pl=1 (accumulate)
    conv(ul_b.w, 0, ul_b.b, xin, nullptr, 2, 0, 0, 1, stUL[0], 64, 4,
         2, 1, 1, 1, 28, 28, 28, 28, 1, 1, 1, 0, 1);

    // ---- up pass (outputs land in the skip stacks; no copies needed) ----
    float* u = stU[0]; float* ul = stUL[0];
    int si = 1, Hs = 28;
    for (int lev = 0; lev < 3; ++lev) {
        for (int r = 0; r < 2; ++r) {
            gated(u, up_u[lev][r], 0, nullptr, nullptr, Hs, stU[si]);
            u = stU[si];
            gated(ul, up_ul[lev][r], 1, u, nullptr, Hs, stUL[si]);
            ul = stUL[si];
            ++si;
        }
        if (lev != 2) {
            int Ho = Hs / 2;
            conv(dsz_u[lev].w, 0, dsz_u[lev].b, u, nullptr, FCH, 0, 0, 0,
                 stU[si], 64, 64 * 6, 2, 3, 1, 1, Hs, Hs, Ho, Ho, 2, 2, 0, 0, 0);
            u = stU[si];
            conv(dsz_ul[lev].w, 0, dsz_ul[lev].b, ul, nullptr, FCH, 0, 0, 0,
                 stUL[si], 64, 64 * 4, 2, 2, 1, 1, Hs, Hs, Ho, Ho, 2, 2, 0, 0, 0);
            ul = stUL[si];
            ++si; Hs = Ho;
        }
    }

    // ---- down pass (pop skips; ping-pong work buffers) ----
    int sp = 7, wi = 0;
    Hs = 7;
    for (int lev = 0; lev < 3; ++lev) {
        for (int r = 0; r < NRDN[lev]; ++r) {
            const float* au = stU[sp]; const float* aul = stUL[sp]; --sp;
            gated(u, dn_u[lev][r], 0, au, nullptr, Hs, wU[wi]);
            float* unew = wU[wi];
            gated(ul, dn_ul[lev][r], 1, unew, aul, Hs, wUL[wi]);
            u = unew; ul = wUL[wi]; wi ^= 1;
        }
        if (lev != 2) {
            int Ho = Hs * 2;
            deconv_kernel<<<ew(NIMG * FCH * Ho * Ho), 256, 0, stream>>>(
                u, usz_u[lev].w, usz_u[lev].b, wU[wi], Hs, Hs, Ho, Ho, 3);
            u = wU[wi];
            deconv_kernel<<<ew(NIMG * FCH * Ho * Ho), 256, 0, stream>>>(
                ul, usz_ul[lev].w, usz_ul[lev].b, wUL[wi], Hs, Hs, Ho, Ho, 2);
            ul = wUL[wi]; wi ^= 1; Hs = Ho;
        }
    }

    // ---- output head ----
    final_kernel<<<ew(NIMG * 784), 256, 0, stream>>>(ul, outP.w, outP.b,
                                                     (float*)d_out);
    (void)in_sizes; (void)n_in; (void)out_size; (void)ws_size;
}
